// MambaBlock_7524782702695
// MI455X (gfx1250) — matrix operand order repaired and checked by reading, not compiled, not run on hardware
//
#include <hip/hip_runtime.h>
#include <hip/hip_bf16.h>
#include <cstdint>

typedef __attribute__((ext_vector_type(16))) _Float16 v16h;
typedef __attribute__((ext_vector_type(8)))  float    v8f;

#define BDIM 4
#define CDIM 256
#define LSEQ 4096
#define BL   (BDIM*LSEQ)
#define DIN  512
#define NXZ  1024
#define DST  16
#define NDBC 48
#define NCHUNK 64
#define CHUNK  64
#define EPSV 1e-5f

// -------------------- f32 -> f16 convert --------------------
__global__ void k_cvt_f16(const float* __restrict__ s, _Float16* __restrict__ d, int n) {
  int i = blockIdx.x * 256 + threadIdx.x;
  if (i < n) d[i] = (_Float16)s[i];
}

// -------------------- LayerNorm over C, emit f16 A-matrix --------------------
__global__ __launch_bounds__(256) void k_layernorm(const float* __restrict__ x,
    const float* __restrict__ w, const float* __restrict__ bia,
    _Float16* __restrict__ xn) {
  int t = blockIdx.x;                    // token = b*L + l
  int c = threadIdx.x;
  int b = t >> 12, l = t & 4095;
  float v = x[((size_t)(b * CDIM + c)) * LSEQ + l];
  __shared__ float red[256];
  __shared__ float mv[2];
  red[c] = v; __syncthreads();
  for (int o = 128; o > 0; o >>= 1) { if (c < o) red[c] += red[c + o]; __syncthreads(); }
  if (c == 0) mv[0] = red[0] * (1.0f / 256.0f);
  __syncthreads();
  float mu = mv[0];
  float dv = v - mu;
  red[c] = dv * dv; __syncthreads();
  for (int o = 128; o > 0; o >>= 1) { if (c < o) red[c] += red[c + o]; __syncthreads(); }
  if (c == 0) mv[1] = red[0] * (1.0f / 256.0f);
  __syncthreads();
  float xv = dv * rsqrtf(mv[1] + EPSV) * w[c] + bia[c];
  xn[(size_t)t * CDIM + c] = (_Float16)xv;
}

// -------------------- WMMA f16 GEMM: C(MxN) = A(MxK) * B(KxN) --------------------
// mode 0: plain f32 store; mode 1: store transposed to (b, c, l) with residual add.
__global__ __launch_bounds__(256) void k_gemm_f16(
    const _Float16* __restrict__ A, const _Float16* __restrict__ B,
    float* __restrict__ C, const float* __restrict__ resid,
    int M, int N, int K, int mode) {
  __shared__ _Float16 sA[64][40];        // [m][k] row-major, padded
  __shared__ _Float16 sB[64][40];        // [n][k] transposed, padded
  int tid  = threadIdx.x;
  int lane = tid & 31;
  int grp  = (lane >> 4) & 1;
  int wv   = tid >> 5;                   // 8 waves
  int mblk = wv & 3;                     // 4 M sub-blocks of 16
  int nbase = (wv >> 2) * 32;            // 2 N strips of 32
  int m0 = blockIdx.y * 64, n0 = blockIdx.x * 64;
  const uint32_t* gA = reinterpret_cast<const uint32_t*>(A);
  const uint32_t* gB = reinterpret_cast<const uint32_t*>(B);
  v8f acc0 = {}, acc1 = {};
  int nk = K >> 5;
  for (int ks = 0; ks < nk; ++ks) {
    int k0 = ks << 5;
    // stage A tile 64x32 halves (1024 dwords)
    for (int i = tid; i < 1024; i += 256) {
      int r = i >> 4, p = i & 15;
      uint32_t dw = gA[(((size_t)(m0 + r)) * K + k0) / 2 + p];
      *reinterpret_cast<uint32_t*>(&sA[r][p * 2]) = dw;
    }
    // stage B tile 32x64 halves transposed into [n][k]
    for (int i = tid; i < 1024; i += 256) {
      int k = i >> 5, np = i & 31;
      uint32_t dw = gB[(((size_t)(k0 + k)) * N + n0) / 2 + np];
      union { uint32_t u; _Float16 h[2]; } pk; pk.u = dw;
      sB[np * 2][k]     = pk.h[0];
      sB[np * 2 + 1][k] = pk.h[1];
    }
    if (ks + 1 < nk) {   // emits global_prefetch_b8 for next K tile
      __builtin_prefetch(&gA[(((size_t)(m0 + (tid & 63))) * K + k0 + 32) / 2], 0, 1);
      __builtin_prefetch(&gB[(((size_t)(k0 + 32 + (tid & 31))) * N + n0) / 2], 0, 1);
    }
    __syncthreads();
    // fragments per ISA 16-bit layouts (wave32)
    union { v16h v; uint32_t u[8]; } fa, fb0, fb1;
    int mr  = mblk * 16 + (lane & 15);
    int nr0 = nbase + (lane & 15);
    int nr1 = nr0 + 16;
#pragma unroll
    for (int j = 0; j < 8; ++j) {
      int kbA = ((j & 4) ? 16 : 0) + 2 * (j & 3) + grp * 8;  // A 16x32 interleave
      int kbB = kbA;                                         // B same K interleave as A
      fa.u[j]  = *reinterpret_cast<const uint32_t*>(&sA[mr][kbA]);
      fb0.u[j] = *reinterpret_cast<const uint32_t*>(&sB[nr0][kbB]);
      fb1.u[j] = *reinterpret_cast<const uint32_t*>(&sB[nr1][kbB]);
    }
    acc0 = __builtin_amdgcn_wmma_f32_16x16x32_f16(false, fa.v, false, fb0.v,
                                                  (short)0, acc0, false, false);
    acc1 = __builtin_amdgcn_wmma_f32_16x16x32_f16(false, fa.v, false, fb1.v,
                                                  (short)0, acc1, false, false);
    __syncthreads();
  }
  // C/D layout: row = vgpr + 8*lanegroup, col = lane&15
#pragma unroll
  for (int v = 0; v < 8; ++v) {
    int row = m0 + mblk * 16 + grp * 8 + v;
    int col = n0 + nbase + (lane & 15);
    if (mode == 0) {
      C[(size_t)row * N + col]      = acc0[v];
      C[(size_t)row * N + col + 16] = acc1[v];
    } else {
      int b = row >> 12, l = row & 4095;
      size_t o0 = ((size_t)(b * CDIM + col)) * LSEQ + l;
      size_t o1 = ((size_t)(b * CDIM + col + 16)) * LSEQ + l;
      C[o0] = acc0[v] + resid[o0];
      C[o1] = acc1[v] + resid[o1];
    }
  }
}

// -------------------- depthwise causal conv1d + bias + SiLU --------------------
__global__ void k_conv_silu(const float* __restrict__ xz, const float* __restrict__ cw,
                            const float* __restrict__ cb, float* __restrict__ uc) {
  int idx = blockIdx.x * 256 + threadIdx.x;     // BL * DIN threads
  int t = idx >> 9, d = idx & 511;
  int l = t & 4095;
  float a = cb[d];
#pragma unroll
  for (int k = 0; k < 4; ++k) {
    int ll = l + k - 3;
    if (ll >= 0) a += cw[d * 4 + k] * xz[((size_t)(t + k - 3)) * NXZ + d];
  }
  uc[(size_t)t * DIN + d] = a / (1.0f + __expf(-a));
}

// -------------------- dbc = uc @ W_x ; dt = softplus(dbc[:,:16] @ W_dt + b_dt) ---
__global__ __launch_bounds__(256) void k_dbc(const float* __restrict__ uc,
    const float* __restrict__ Wx, const float* __restrict__ Wdt,
    const float* __restrict__ bdt, float* __restrict__ dt,
    float* __restrict__ Bm, float* __restrict__ Cm) {
  int t = blockIdx.x, tid = threadIdx.x;
  __shared__ float su[DIN];
  __shared__ float sd[NDBC];
  su[tid]       = uc[(size_t)t * DIN + tid];
  su[tid + 256] = uc[(size_t)t * DIN + tid + 256];
  __syncthreads();
  if (tid < NDBC) {
    float a = 0.0f;
    for (int k = 0; k < DIN; ++k) a += su[k] * Wx[k * NDBC + tid];
    sd[tid] = a;
  }
  __syncthreads();
  for (int d = tid; d < DIN; d += 256) {
    float a = bdt[d];
#pragma unroll
    for (int r = 0; r < 16; ++r) a += sd[r] * Wdt[r * DIN + d];
    dt[(size_t)t * DIN + d] = (a > 20.0f) ? a : log1pf(__expf(a));
  }
  if (tid < DST)            Bm[(size_t)t * DST + tid]        = sd[DST + tid];
  else if (tid < 2 * DST)   Cm[(size_t)t * DST + (tid - DST)] = sd[2 * DST + (tid - DST)];
}

// -------------------- scan pass 1: per-chunk cumdecay + local state ------------
__global__ __launch_bounds__(512) void k_scan1(const float* __restrict__ dt,
    const float* __restrict__ uc, const float* __restrict__ Bm,
    const float* __restrict__ A_log, float* __restrict__ Pc, float* __restrict__ Lc) {
  int b = blockIdx.x >> 6, j = blockIdx.x & 63;
  int d = threadIdx.x;
  float Ae[DST], h[DST], P[DST];
#pragma unroll
  for (int s = 0; s < DST; ++s) {
    Ae[s] = -__expf(A_log[d * DST + s]);
    h[s] = 0.0f; P[s] = 1.0f;
  }
  __shared__ float sB[DST];
  for (int q = 0; q < CHUNK; ++q) {
    int t = (b << 12) | (j * CHUNK + q);
    if (d < DST) sB[d] = Bm[(size_t)t * DST + d];
    __syncthreads();
    float dtv = dt[(size_t)t * DIN + d];
    float du  = dtv * uc[(size_t)t * DIN + d];
#pragma unroll
    for (int s = 0; s < DST; ++s) {
      float e = __expf(dtv * Ae[s]);
      h[s] = e * h[s] + du * sB[s];
      P[s] *= e;
    }
    __syncthreads();
  }
  size_t o = (((size_t)(b * NCHUNK + j)) * DIN + d) * DST;
#pragma unroll
  for (int s = 0; s < DST; ++s) { Pc[o + s] = P[s]; Lc[o + s] = h[s]; }
}

// -------------------- scan pass 2: sequential carry over chunks ----------------
__global__ void k_scan2(const float* __restrict__ Pc, const float* __restrict__ Lc,
                        float* __restrict__ Hin) {
  int idx = blockIdx.x * 256 + threadIdx.x;     // (b, d, s) = 4*512*16
  if (idx >= BDIM * DIN * DST) return;
  int s = idx & 15, d = (idx >> 4) & 511, b = idx >> 13;
  float h = 0.0f;
  for (int j = 0; j < NCHUNK; ++j) {
    size_t o = (((size_t)(b * NCHUNK + j)) * DIN + d) * DST + s;
    Hin[o] = h;
    h = Pc[o] * h + Lc[o];
  }
}

// -------------------- scan pass 3: replay with carry, fuse D/silu(z), f16 out --
__global__ __launch_bounds__(512) void k_scan3(const float* __restrict__ dt,
    const float* __restrict__ uc, const float* __restrict__ xz,
    const float* __restrict__ Bm, const float* __restrict__ Cm,
    const float* __restrict__ A_log, const float* __restrict__ Dp,
    const float* __restrict__ Hin, _Float16* __restrict__ yact) {
  int b = blockIdx.x >> 6, j = blockIdx.x & 63;
  int d = threadIdx.x;
  float Ae[DST], h[DST];
  size_t oh = (((size_t)(b * NCHUNK + j)) * DIN + d) * DST;
#pragma unroll
  for (int s = 0; s < DST; ++s) {
    Ae[s] = -__expf(A_log[d * DST + s]);
    h[s] = Hin[oh + s];
  }
  float Dpar = Dp[d];
  __shared__ float sB[DST], sC[DST];
  for (int q = 0; q < CHUNK; ++q) {
    int t = (b << 12) | (j * CHUNK + q);
    if (d < DST)          sB[d]        = Bm[(size_t)t * DST + d];
    else if (d < 2 * DST) sC[d - DST]  = Cm[(size_t)t * DST + (d - DST)];
    __syncthreads();
    float dtv = dt[(size_t)t * DIN + d];
    float ucv = uc[(size_t)t * DIN + d];
    float zv  = xz[(size_t)t * NXZ + DIN + d];
    float du = dtv * ucv;
    float y = 0.0f;
#pragma unroll
    for (int s = 0; s < DST; ++s) {
      float e = __expf(dtv * Ae[s]);
      h[s] = e * h[s] + du * sB[s];
      y += h[s] * sC[s];
    }
    y += ucv * Dpar;
    y *= zv / (1.0f + __expf(-zv));
    yact[(size_t)t * DIN + d] = (_Float16)y;
    __syncthreads();
  }
}

// -------------------- host side --------------------
extern "C" void kernel_launch(void* const* d_in, const int* in_sizes, int n_in,
                              void* d_out, int out_size, void* d_ws, size_t ws_size,
                              hipStream_t stream) {
  const float* x      = (const float*)d_in[0];
  const float* norm_w = (const float*)d_in[1];
  const float* norm_b = (const float*)d_in[2];
  const float* W_in   = (const float*)d_in[3];
  const float* conv_w = (const float*)d_in[4];
  const float* conv_b = (const float*)d_in[5];
  const float* W_x    = (const float*)d_in[6];
  const float* W_dt   = (const float*)d_in[7];
  const float* b_dt   = (const float*)d_in[8];
  const float* A_log  = (const float*)d_in[9];
  const float* D_par  = (const float*)d_in[10];
  const float* W_out  = (const float*)d_in[11];
  float* out = (float*)d_out;

  char* ws = (char*)d_ws;
  size_t off = 0;
  _Float16* Win_h  = (_Float16*)(ws + off); off += (size_t)CDIM * NXZ * 2;   // 0.5 MB
  _Float16* Wout_h = (_Float16*)(ws + off); off += (size_t)DIN * CDIM * 2;   // 0.25 MB
  _Float16* xn_h   = (_Float16*)(ws + off); off += (size_t)BL * CDIM * 2;    // 8.4 MB
  float*    xz     = (float*)(ws + off);    off += (size_t)BL * NXZ * 4;     // 67 MB
  float*    uc     = (float*)(ws + off);    off += (size_t)BL * DIN * 4;     // 33.5 MB
  float*    dtb    = (float*)(ws + off);    off += (size_t)BL * DIN * 4;     // 33.5 MB
  float*    Bmv    = (float*)(ws + off);    off += (size_t)BL * DST * 4;     // 1 MB
  float*    Cmv    = (float*)(ws + off);    off += (size_t)BL * DST * 4;     // 1 MB
  _Float16* yact   = (_Float16*)(ws + off); off += (size_t)BL * DIN * 2;     // 16.8 MB
  float*    Pc     = (float*)(ws + off);    off += (size_t)BDIM * NCHUNK * DIN * DST * 4; // 8.4 MB
  float*    Lc     = (float*)(ws + off);    off += (size_t)BDIM * NCHUNK * DIN * DST * 4; // 8.4 MB
  float*    Hin    = (float*)(ws + off);    off += (size_t)BDIM * NCHUNK * DIN * DST * 4; // 8.4 MB
  (void)off; (void)in_sizes; (void)n_in; (void)out_size; (void)ws_size;

  k_cvt_f16<<<(CDIM * NXZ + 255) / 256, 256, 0, stream>>>(W_in, Win_h, CDIM * NXZ);
  k_cvt_f16<<<(DIN * CDIM + 255) / 256, 256, 0, stream>>>(W_out, Wout_h, DIN * CDIM);
  k_layernorm<<<BL, 256, 0, stream>>>(x, norm_w, norm_b, xn_h);

  dim3 g1(NXZ / 64, BL / 64);
  k_gemm_f16<<<g1, 256, 0, stream>>>(xn_h, Win_h, xz, nullptr, BL, NXZ, CDIM, 0);

  k_conv_silu<<<(BL * DIN) / 256, 256, 0, stream>>>(xz, conv_w, conv_b, uc);
  k_dbc<<<BL, 256, 0, stream>>>(uc, W_x, W_dt, b_dt, dtb, Bmv, Cmv);

  k_scan1<<<BDIM * NCHUNK, 512, 0, stream>>>(dtb, uc, Bmv, A_log, Pc, Lc);
  k_scan2<<<(BDIM * DIN * DST + 255) / 256, 256, 0, stream>>>(Pc, Lc, Hin);
  k_scan3<<<BDIM * NCHUNK, 512, 0, stream>>>(dtb, uc, xz, Bmv, Cmv, A_log, D_par, Hin, yact);

  dim3 g2(CDIM / 64, BL / 64);
  k_gemm_f16<<<g2, 256, 0, stream>>>(yact, Wout_h, out, x, BL, CDIM, DIN, 1);
}
